// MultiHeadAttention_62234076119819
// MI455X (gfx1250) — compile-verified, matrix operand order checked by reading, not run on hardware
//
#include <hip/hip_runtime.h>
#include <stdint.h>

#define EMBED 1024
#define HEADS 16
#define HD    64
#define SEQ   2048
#define BATCH 4
#define ROWS  (BATCH * SEQ)   // 8192

typedef __attribute__((ext_vector_type(16))) __bf16 v16bf;
typedef __attribute__((ext_vector_type(8)))  float  v8f;
typedef __attribute__((ext_vector_type(4)))  unsigned int u32x4;

union BF16x16 { u32x4 u[2]; v16bf v; };   // 8 VGPRs = 16 bf16 (one WMMA A/B operand)

__device__ __forceinline__ unsigned short f2bf(float f) {
    unsigned u = __float_as_uint(f);
    u += 0x7FFFu + ((u >> 16) & 1u);      // round-to-nearest-even
    return (unsigned short)(u >> 16);
}

__device__ __forceinline__ v8f wmma_bf(const BF16x16& a, const BF16x16& b, v8f c) {
    return __builtin_amdgcn_wmma_f32_16x16x32_bf16(false, a.v, false, b.v,
                                                   (short)0, c, false, false);
}

__device__ __forceinline__ float rowmax16(float v) {  // reduce across 16-lane half
    v = fmaxf(v, __shfl_xor(v, 1, 32));
    v = fmaxf(v, __shfl_xor(v, 2, 32));
    v = fmaxf(v, __shfl_xor(v, 4, 32));
    v = fmaxf(v, __shfl_xor(v, 8, 32));
    return v;
}
__device__ __forceinline__ float rowsum16(float v) {
    v += __shfl_xor(v, 1, 32);
    v += __shfl_xor(v, 2, 32);
    v += __shfl_xor(v, 4, 32);
    v += __shfl_xor(v, 8, 32);
    return v;
}

// ---- CDNA5 async global->LDS copy (ASYNCcnt path), 16 bytes per lane ----
__device__ __forceinline__ unsigned lds_off(const void* p) {
    // LDS aperture: flat addr[31:0] is the group-relative LDS byte offset
    return (unsigned)(unsigned long long)p;
}
__device__ __forceinline__ void async_copy16(unsigned ldsOff, const unsigned short* g) {
    asm volatile("global_load_async_to_lds_b128 %0, %1, off"
                 :: "v"(ldsOff), "v"(g) : "memory");
}
__device__ __forceinline__ void wait_async0() {
    asm volatile("s_wait_asynccnt 0x0" ::: "memory");
}

// ---------------- LayerNorm (fp32 -> bf16) -----------------
__global__ __launch_bounds__(256) void ln_kernel(const float* __restrict__ x,
                                                 const float* __restrict__ gamma,
                                                 const float* __restrict__ beta,
                                                 unsigned short* __restrict__ xn) {
    __shared__ float red[16];
    const int row = blockIdx.x;
    const int t = threadIdx.x;                 // 256 threads * 4 elems = 1024
    const float* xr = x + (size_t)row * EMBED;
    float4 v = reinterpret_cast<const float4*>(xr)[t];
    float s  = v.x + v.y + v.z + v.w;
    float s2 = v.x * v.x + v.y * v.y + v.z * v.z + v.w * v.w;
    for (int m = 16; m; m >>= 1) { s += __shfl_xor(s, m, 32); s2 += __shfl_xor(s2, m, 32); }
    if ((t & 31) == 0) { red[t >> 5] = s; red[8 + (t >> 5)] = s2; }
    __syncthreads();
    s = 0.f; s2 = 0.f;
    #pragma unroll
    for (int i = 0; i < 8; i++) { s += red[i]; s2 += red[8 + i]; }
    const float mu   = s * (1.0f / EMBED);
    const float var  = s2 * (1.0f / EMBED) - mu * mu;
    const float rstd = rsqrtf(var + 1e-6f);
    const int c = t * 4;
    unsigned short b0 = f2bf((v.x - mu) * rstd * gamma[c + 0] + beta[c + 0]);
    unsigned short b1 = f2bf((v.y - mu) * rstd * gamma[c + 1] + beta[c + 1]);
    unsigned short b2 = f2bf((v.z - mu) * rstd * gamma[c + 2] + beta[c + 2]);
    unsigned short b3 = f2bf((v.w - mu) * rstd * gamma[c + 3] + beta[c + 3]);
    uint2 packed;
    packed.x = (unsigned)b0 | ((unsigned)b1 << 16);
    packed.y = (unsigned)b2 | ((unsigned)b3 << 16);
    reinterpret_cast<uint2*>(xn + (size_t)row * EMBED)[t] = packed;
}

// ---------------- fp32 -> bf16 weight convert -----------------
__global__ __launch_bounds__(256) void cvt_bf16(const float* __restrict__ in,
                                                unsigned short* __restrict__ out, int n4) {
    const int i = blockIdx.x * blockDim.x + threadIdx.x;
    if (i >= n4) return;
    float4 v = reinterpret_cast<const float4*>(in)[i];
    uint2 packed;
    packed.x = (unsigned)f2bf(v.x) | ((unsigned)f2bf(v.y) << 16);
    packed.y = (unsigned)f2bf(v.z) | ((unsigned)f2bf(v.w) << 16);
    reinterpret_cast<uint2*>(out)[i] = packed;
}

// ---------------- WMMA GEMM: Y = X @ W^T + b ------------------
// Workgroup tile 128x64: 8 waves stacked in M, sharing a 64x32 B-chunk staged
// into double-buffered LDS via async copies. A is register-prefetched; all B
// operands are batch-loaded from LDS before the WMMA burst.
// mode 0: Q bf16 [B,H,S,hd] scaled by 1/sqrt(hd)
// mode 1: K bf16 [B,H,S,hd]
// mode 2: V bf16 [B,H,hd,S]
// mode 3: fp32 row-major out
__global__ __launch_bounds__(256) void gemm_wmma(const unsigned short* __restrict__ X,
                                                 const unsigned short* __restrict__ W,
                                                 const float* __restrict__ bias,
                                                 unsigned short* __restrict__ outb,
                                                 float* __restrict__ outf,
                                                 int mode) {
    __shared__ unsigned short Bsh[2][64][32];   // 2 x 4KB
    const int K = EMBED, N = EMBED;
    const int t    = threadIdx.x;
    const int win  = t >> 5;
    const int lane = t & 31;
    const int mblk = blockIdx.x >> 4;           // 64 M-blocks of 128 rows
    const int n0   = (blockIdx.x & 15) * 64;    // 16 N-blocks of 64 cols
    const int m0   = mblk * 128 + win * 16;
    const int half = lane >> 4, lr = lane & 15;
    const int kb   = half * 8;

    // staging coordinates: one 16B chunk per thread per K-step
    const int srow = t >> 2;                    // 0..63 (B row = N column index)
    const int sch  = (t & 3) * 8;               // elem offset 0,8,16,24

    const unsigned short* arow = X + (size_t)(m0 + lr) * K;
    v8f acc[4];
    #pragma unroll
    for (int nt = 0; nt < 4; nt++)
        #pragma unroll
        for (int r = 0; r < 8; r++) acc[nt][r] = 0.f;

    // prologue: stage k0=0 B-chunk, load first A chunk
    async_copy16(lds_off(&Bsh[0][srow][sch]),
                 W + (size_t)(n0 + srow) * K + sch);
    BF16x16 a_cur, a_nxt;
    a_cur.u[0] = *reinterpret_cast<const u32x4*>(arow + kb);
    a_cur.u[1] = *reinterpret_cast<const u32x4*>(arow + kb + 16);

    int buf = 0;
    for (int k0 = 0; k0 < K; k0 += 32) {
        wait_async0();
        __syncthreads();
        if (k0 + 32 < K) {
            async_copy16(lds_off(&Bsh[buf ^ 1][srow][sch]),
                         W + (size_t)(n0 + srow) * K + k0 + 32 + sch);
            a_nxt.u[0] = *reinterpret_cast<const u32x4*>(arow + k0 + 32 + kb);
            a_nxt.u[1] = *reinterpret_cast<const u32x4*>(arow + k0 + 32 + kb + 16);
        }
        // batch all 8 ds_load_b128 first so WMMAs overlap LDS returns
        BF16x16 b[4];
        #pragma unroll
        for (int nt = 0; nt < 4; nt++) {
            const int row = nt * 16 + lr;
            b[nt].u[0] = *reinterpret_cast<const u32x4*>(&Bsh[buf][row][kb]);
            b[nt].u[1] = *reinterpret_cast<const u32x4*>(&Bsh[buf][row][kb + 16]);
        }
        #pragma unroll
        for (int nt = 0; nt < 4; nt++)
            acc[nt] = wmma_bf(a_cur, b[nt], acc[nt]);
        if (k0 + 32 < K) a_cur = a_nxt;
        buf ^= 1;
    }

    #pragma unroll
    for (int nt = 0; nt < 4; nt++) {
        const int Ncol = n0 + nt * 16 + lr;
        const float bv = bias[Ncol];
        #pragma unroll
        for (int r = 0; r < 8; r++) {
            const int Mrow = m0 + r + 8 * half;
            float val = acc[nt][r] + bv;
            if (mode == 3) {
                outf[(size_t)Mrow * N + Ncol] = val;
            } else {
                if (mode == 0) val *= 0.125f;          // 1/sqrt(64)
                const int b_ = Mrow >> 11, s = Mrow & (SEQ - 1);
                const int h = Ncol >> 6,  d = Ncol & (HD - 1);
                size_t idx;
                if (mode == 2) idx = (((size_t)(b_ * HEADS + h)) * HD + d) * SEQ + s;
                else           idx = (((size_t)(b_ * HEADS + h)) * SEQ + s) * HD + d;
                outb[idx] = f2bf(val);
            }
        }
    }
}

// ---------------- Flash attention ----------------
// All 8 waves of a block share one (b,h); K/V 32-key chunks are staged once
// into double-buffered LDS via async copies and consumed by all 8 q-tiles.
__global__ __launch_bounds__(256) void attn_wmma(const unsigned short* __restrict__ Q,
                                                 const unsigned short* __restrict__ Kd,
                                                 const unsigned short* __restrict__ Vt,
                                                 unsigned short* __restrict__ ctx) {
    __shared__ unsigned short Ksh[2][32][64];   // 2 x 4KB: 32 keys x 64 hd
    __shared__ unsigned short Vsh[2][64][32];   // 2 x 4KB: 64 hd x 32 keys
    __shared__ unsigned short Psh[8][16 * 32];  // per-wave P re-stripe tile
    const int t    = threadIdx.x;
    const int win  = t >> 5;
    const int lane = t & 31;
    const int bh   = blockIdx.x >> 4;                  // 0..63
    const int qt   = (blockIdx.x & 15) * 8 + win;      // 0..127
    const int b = bh >> 4, h = bh & 15;
    const int half = lane >> 4, lr = lane & 15;
    const int kb = half * 8;

    const unsigned short* Qb = Q  + (size_t)bh * SEQ * HD;
    const unsigned short* Kb = Kd + (size_t)bh * SEQ * HD;
    const unsigned short* Vb = Vt + (size_t)bh * HD * SEQ;

    // staging coords: each thread moves one 16B chunk of K and one of V per step
    const int krw = t >> 3, kch = (t & 7) * 8;   // K: 32 rows x 64 elems
    const int vrw = t >> 2, vch = (t & 3) * 8;   // V: 64 rows x 32 elems

    // Q tile 16x64 in two A-operands
    const unsigned short* qrow = Qb + (size_t)(qt * 16 + lr) * HD;
    BF16x16 q0, q1;
    q0.u[0] = *reinterpret_cast<const u32x4*>(qrow + kb);
    q0.u[1] = *reinterpret_cast<const u32x4*>(qrow + kb + 16);
    q1.u[0] = *reinterpret_cast<const u32x4*>(qrow + kb + 32);
    q1.u[1] = *reinterpret_cast<const u32x4*>(qrow + kb + 48);

    v8f acc[4];
    float rmax[8], rsum[8];
    #pragma unroll
    for (int r = 0; r < 8; r++) {
        rmax[r] = -3.0e38f; rsum[r] = 0.f;
        #pragma unroll
        for (int nt = 0; nt < 4; nt++) acc[nt][r] = 0.f;
    }

    // prologue staging for key chunk 0
    async_copy16(lds_off(&Ksh[0][krw][kch]), Kb + (size_t)krw * HD + kch);
    async_copy16(lds_off(&Vsh[0][vrw][vch]), Vb + (size_t)vrw * SEQ + vch);

    int buf = 0;
    for (int key0 = 0; key0 < SEQ; key0 += 32) {
        wait_async0();
        __syncthreads();
        if (key0 + 32 < SEQ) {
            async_copy16(lds_off(&Ksh[buf ^ 1][krw][kch]),
                         Kb + (size_t)(key0 + 32 + krw) * HD + kch);
            async_copy16(lds_off(&Vsh[buf ^ 1][vrw][vch]),
                         Vb + (size_t)vrw * SEQ + key0 + 32 + vch);
        }

        // batch all 8 K-operand ds loads, then the 4 score WMMAs
        BF16x16 kb00, kb01, kb10, kb11;
        kb00.u[0] = *reinterpret_cast<const u32x4*>(&Ksh[buf][lr][kb]);
        kb00.u[1] = *reinterpret_cast<const u32x4*>(&Ksh[buf][lr][kb + 16]);
        kb01.u[0] = *reinterpret_cast<const u32x4*>(&Ksh[buf][lr][kb + 32]);
        kb01.u[1] = *reinterpret_cast<const u32x4*>(&Ksh[buf][lr][kb + 48]);
        kb10.u[0] = *reinterpret_cast<const u32x4*>(&Ksh[buf][16 + lr][kb]);
        kb10.u[1] = *reinterpret_cast<const u32x4*>(&Ksh[buf][16 + lr][kb + 16]);
        kb11.u[0] = *reinterpret_cast<const u32x4*>(&Ksh[buf][16 + lr][kb + 32]);
        kb11.u[1] = *reinterpret_cast<const u32x4*>(&Ksh[buf][16 + lr][kb + 48]);

        v8f s0, s1;
        #pragma unroll
        for (int r = 0; r < 8; r++) { s0[r] = 0.f; s1[r] = 0.f; }
        s0 = wmma_bf(q0, kb00, s0);
        s0 = wmma_bf(q1, kb01, s0);
        s1 = wmma_bf(q0, kb10, s1);
        s1 = wmma_bf(q1, kb11, s1);

        // online softmax update per C-layout row (M = r + 8*half)
        #pragma unroll
        for (int r = 0; r < 8; r++) {
            float cm = rowmax16(fmaxf(s0[r], s1[r]));
            float nm = fmaxf(rmax[r], cm);
            float sc = __expf(rmax[r] - nm);
            rmax[r] = nm;
            float p0 = __expf(s0[r] - nm);
            float p1 = __expf(s1[r] - nm);
            rsum[r] = rsum[r] * sc + rowsum16(p0 + p1);
            acc[0][r] *= sc; acc[1][r] *= sc; acc[2][r] *= sc; acc[3][r] *= sc;
            const int row = r + 8 * half;
            Psh[win][row * 32 + lr]      = f2bf(p0);
            Psh[win][row * 32 + 16 + lr] = f2bf(p1);
        }

        // P (16x32) @ V (32x64); batch P + all V-operand loads, then 4 WMMAs
        BF16x16 ap;
        ap.u[0] = *reinterpret_cast<const u32x4*>(&Psh[win][lr * 32 + kb]);
        ap.u[1] = *reinterpret_cast<const u32x4*>(&Psh[win][lr * 32 + kb + 16]);
        BF16x16 bv[4];
        #pragma unroll
        for (int nt = 0; nt < 4; nt++) {
            bv[nt].u[0] = *reinterpret_cast<const u32x4*>(&Vsh[buf][nt * 16 + lr][kb]);
            bv[nt].u[1] = *reinterpret_cast<const u32x4*>(&Vsh[buf][nt * 16 + lr][kb + 16]);
        }
        #pragma unroll
        for (int nt = 0; nt < 4; nt++)
            acc[nt] = wmma_bf(ap, bv[nt], acc[nt]);
        buf ^= 1;
    }

    // normalize and write ctx as bf16 row-major [B*S, EMBED]
    #pragma unroll
    for (int r = 0; r < 8; r++) {
        const float inv = 1.0f / rsum[r];
        const int Mrow = b * SEQ + qt * 16 + r + 8 * half;
        #pragma unroll
        for (int nt = 0; nt < 4; nt++) {
            ctx[(size_t)Mrow * EMBED + h * HD + nt * 16 + lr] = f2bf(acc[nt][r] * inv);
        }
    }
}

// ---------------- host-side launch ----------------
extern "C" void kernel_launch(void* const* d_in, const int* in_sizes, int n_in,
                              void* d_out, int out_size, void* d_ws, size_t ws_size,
                              hipStream_t stream) {
    const float* x  = (const float*)d_in[0];
    const float* g  = (const float*)d_in[1];
    const float* bt = (const float*)d_in[2];
    const float* Wq = (const float*)d_in[3];  const float* bq = (const float*)d_in[4];
    const float* Wk = (const float*)d_in[5];  const float* bk = (const float*)d_in[6];
    const float* Wv = (const float*)d_in[7];  const float* bv = (const float*)d_in[8];
    const float* Wo = (const float*)d_in[9];  const float* bo = (const float*)d_in[10];

    char* ws = (char*)d_ws;
    const size_t MB = 1u << 20;
    unsigned short* xn = (unsigned short*)(ws);             // 16 MiB; later reused as ctx
    unsigned short* wb = (unsigned short*)(ws + 16 * MB);   //  8 MiB (4 bf16 weight mats)
    unsigned short* Qb = (unsigned short*)(ws + 24 * MB);   // 16 MiB
    unsigned short* Kb = (unsigned short*)(ws + 40 * MB);   // 16 MiB
    unsigned short* Vt = (unsigned short*)(ws + 56 * MB);   // 16 MiB (total 72 MiB)

    const int wn = EMBED * EMBED;

    ln_kernel<<<ROWS, 256, 0, stream>>>(x, g, bt, xn);

    cvt_bf16<<<wn / 4 / 256, 256, 0, stream>>>(Wq, wb + 0 * (size_t)wn, wn / 4);
    cvt_bf16<<<wn / 4 / 256, 256, 0, stream>>>(Wk, wb + 1 * (size_t)wn, wn / 4);
    cvt_bf16<<<wn / 4 / 256, 256, 0, stream>>>(Wv, wb + 2 * (size_t)wn, wn / 4);
    cvt_bf16<<<wn / 4 / 256, 256, 0, stream>>>(Wo, wb + 3 * (size_t)wn, wn / 4);

    // 64 M-blocks x 16 N-blocks = 1024 workgroups of 8 waves
    gemm_wmma<<<1024, 256, 0, stream>>>(xn, wb + 0 * (size_t)wn, bq, Qb, nullptr, 0);
    gemm_wmma<<<1024, 256, 0, stream>>>(xn, wb + 1 * (size_t)wn, bk, Kb, nullptr, 1);
    gemm_wmma<<<1024, 256, 0, stream>>>(xn, wb + 2 * (size_t)wn, bv, Vt, nullptr, 2);

    // 64 (b,h) x 16 blocks of 8 q-tiles; ctx reuses the xn buffer
    attn_wmma<<<1024, 256, 0, stream>>>(Qb, Kb, Vt, xn);

    gemm_wmma<<<1024, 256, 0, stream>>>(xn, wb + 3 * (size_t)wn, bo, nullptr, (float*)d_out, 3);
}
